// GraphSAGE_26285199852185
// MI455X (gfx1250) — matrix operand order repaired and checked by reading, not compiled, not run on hardware
//
#include <hip/hip_runtime.h>

// ---------------------------------------------------------------------------
// GraphSAGE (N=10000, K=10, D=H=256) on MI455X / gfx1250.
// out = act(src @ Wself + mean(nbr,K) @ Wnbr), applied 3 times.
// HBM-bound (~1.25 GB @ 23.3 TB/s): producer waves stream activations +
// fused neighbor mean into double-buffered LDS tiles while consumer waves
// run a software-pipelined v_wmma_f32_16x16x32_bf16 loop against
// LDS-resident bf16 weights.
// ---------------------------------------------------------------------------

typedef __attribute__((ext_vector_type(16))) __bf16        v16bf;
typedef __attribute__((ext_vector_type(8)))  float         v8f;
typedef __attribute__((ext_vector_type(4)))  float         vf4;
typedef __attribute__((ext_vector_type(2)))  unsigned int  vu2;

union Frag {
    v16bf v;
    uint4 q[2];   // 2 x 16B = 32B
};

#define DHID 256            // feature dim (D == H == 256)
#define KNBR 10             // sampled neighbors per node
#define LPAD 8              // LDS row padding (bf16 elems) -> conflict-free b128
#define LSTR (DHID + LPAD)  // 264
#define NTHREADS 384        // 12 waves: 8 producer + 4 consumer
#define NPROD 256           // producer thread count (waves 0..7)

__device__ __forceinline__ unsigned int bf16_rne(float f) {
    unsigned int u = __builtin_bit_cast(unsigned int, f);
    return (u + 0x7FFFu + ((u >> 16) & 1u)) >> 16;
}
__device__ __forceinline__ unsigned int bf16_pk2(float a, float b) {
    return bf16_rne(a) | (bf16_rne(b) << 16);
}

// non-temporal 4-element loads (activation streams are single-use)
__device__ __forceinline__ vf4 load4(const float* p, size_t i) {
    return __builtin_nontemporal_load(reinterpret_cast<const vf4*>(p + i));
}
__device__ __forceinline__ vf4 load4(const unsigned short* p, size_t i) {
    vu2 u = __builtin_nontemporal_load(reinterpret_cast<const vu2*>(p + i));
    vf4 r;
    r.x = __builtin_bit_cast(float, u.x << 16);
    r.y = __builtin_bit_cast(float, u.x & 0xFFFF0000u);
    r.z = __builtin_bit_cast(float, u.y << 16);
    r.w = __builtin_bit_cast(float, u.y & 0xFFFF0000u);
    return r;
}

__device__ __forceinline__ void store1(float* p, float v) { *p = v; }
__device__ __forceinline__ void store1(unsigned short* p, float v) {
    *p = (unsigned short)bf16_rne(v);
}

// A fragment (16x32 bf16): lane-half lh holds K chunks [ks*32+lh*8,+8) and +16.
__device__ __forceinline__ void load_afrag(Frag& f, const unsigned short* base, int off) {
    f.q[0] = *reinterpret_cast<const uint4*>(base + off);
    f.q[1] = *reinterpret_cast<const uint4*>(base + off + 16);
}
// B fragment (32x16 bf16): lane-half lh holds K chunks [ks*32+lh*8,+8) and +16 at fixed col.
__device__ __forceinline__ void load_bfrag(Frag& f, const unsigned short* base, int off) {
    f.q[0] = *reinterpret_cast<const uint4*>(base + off);
    f.q[1] = *reinterpret_cast<const uint4*>(base + off + 16);
}

// Producer staging: 16 src rows -> bf16 LDS tile, fused mean of 10 neighbor
// rows -> bf16 LDS tile. 1024 float4 tasks over NPROD threads.
template <typename TIn>
__device__ __forceinline__ void stage_block(
    const TIn* __restrict__ src, const TIn* __restrict__ nbr,
    int row0, int tid, unsigned short* aS, unsigned short* aN)
{
    uint2* s2 = reinterpret_cast<uint2*>(aS);
    uint2* n2 = reinterpret_cast<uint2*>(aN);
    for (int i = tid; i < 16 * 64; i += NPROD) {
        const int m = i >> 6;          // row within block
        const int p = i & 63;          // group of 4 columns
        const size_t r = (size_t)(row0 + m);
        vf4 s = load4(src, r * DHID + 4 * p);
        s2[m * (LSTR / 4) + p] = make_uint2(bf16_pk2(s.x, s.y), bf16_pk2(s.z, s.w));
        float a0 = 0.f, a1 = 0.f, a2 = 0.f, a3 = 0.f;
#pragma unroll
        for (int k = 0; k < KNBR; ++k) {
            vf4 t = load4(nbr, (r * KNBR + k) * DHID + 4 * p);
            a0 += t.x; a1 += t.y; a2 += t.z; a3 += t.w;
        }
        n2[m * (LSTR / 4) + p] =
            make_uint2(bf16_pk2(a0 * 0.1f, a1 * 0.1f), bf16_pk2(a2 * 0.1f, a3 * 0.1f));
    }
}

template <typename TIn, typename TOut>
__global__ __launch_bounds__(NTHREADS) void sage_wmma_kernel(
    const TIn*  __restrict__ src,    // [n, 256]
    const TIn*  __restrict__ nbr,    // [n*10, 256] (row m neighbors = rows 10m..10m+9)
    const float* __restrict__ Wself, // [256, 256] row-major (K x N)
    const float* __restrict__ Wnbr,  // [256, 256]
    TOut* __restrict__ out,          // [n, 256]
    int nrows, int do_relu)
{
    // LDS: transposed bf16 weights + double-buffered A tiles (~297 KB)
    __shared__ __align__(16) unsigned short sWs[DHID * LSTR];
    __shared__ __align__(16) unsigned short sWn[DHID * LSTR];
    __shared__ __align__(16) unsigned short sA[2][2][16 * LSTR];

    const int tid = threadIdx.x;

    // Stage both weight matrices once (coalesced fp32 read -> bf16, transposed).
    for (int i = tid; i < DHID * DHID; i += NTHREADS) {
        const int k = i >> 8;
        const int c = i & 255;
        sWs[c * LSTR + k] = (unsigned short)bf16_rne(Wself[i]);
        sWn[c * LSTR + k] = (unsigned short)bf16_rne(Wnbr[i]);
    }

    const int lane = tid & 31;
    const int lm   = lane & 15;     // M (A) / N (B,C) index within a tile
    const int lh   = lane >> 4;     // lane-half selects K / M sub-range
    const int nblk = nrows >> 4;    // 16-row blocks (both N are multiples of 16)
    // wave id as a scalar -> pure s_cbranch control flow, no EXEC games
    const int wvu  = __builtin_amdgcn_readfirstlane(tid >> 5);
    const bool producer = (wvu < 8);   // waves 0..7 stream memory, 8..11 do WMMA
    const int cwv = wvu - 8;           // consumer wave id (0..3), 4 col tiles each

    // Prologue: producers stage the first block into buffer 0.
    if (producer && (int)blockIdx.x < nblk)
        stage_block(src, nbr, blockIdx.x << 4, tid, sA[0][0], sA[0][1]);
    __syncthreads();

    int par = 0;
    for (int blk = blockIdx.x; blk < nblk; blk += gridDim.x) {
        if (producer) {
            // Stream block blk+stride into the other buffer while consumers
            // compute. Independent DScnt chain from the consumer waves.
            const int nb = blk + gridDim.x;
            if (nb < nblk)
                stage_block(src, nbr, nb << 4, tid, sA[par ^ 1][0], sA[par ^ 1][1]);
        } else {
            const unsigned short* Ap[2] = { sA[par][0], sA[par][1] };  // self, nbr
            const unsigned short* Wp[2] = { sWs, sWn };
            v8f acc[4] = {{}, {}, {}, {}};

            // column bases for this consumer wave's 4 tiles
            int wcol[4];
#pragma unroll
            for (int t = 0; t < 4; ++t)
                wcol[t] = ((cwv * 4 + t) * 16 + lm) * LSTR;
            const int arow = lm * LSTR;

            // 16 unified k-steps (8 self + 8 nbr), software-pipelined one step
            // ahead so ds_loads for step s+1 issue before the wmmas of step s.
            Frag a, b0, b1, b2, b3;
            load_afrag(a,  Ap[0], arow + lh * 8);
            load_bfrag(b0, Wp[0], wcol[0] + lh * 8);
            load_bfrag(b1, Wp[0], wcol[1] + lh * 8);
            load_bfrag(b2, Wp[0], wcol[2] + lh * 8);
            load_bfrag(b3, Wp[0], wcol[3] + lh * 8);
#pragma unroll
            for (int s = 0; s < 16; ++s) {
                Frag an, b0n, b1n, b2n, b3n;
                if (s < 15) {
                    const int sn   = s + 1;
                    const int half = sn >> 3;
                    const int ks   = sn & 7;
                    load_afrag(an,  Ap[half], arow + ks * 32 + lh * 8);
                    load_bfrag(b0n, Wp[half], wcol[0] + ks * 32 + lh * 8);
                    load_bfrag(b1n, Wp[half], wcol[1] + ks * 32 + lh * 8);
                    load_bfrag(b2n, Wp[half], wcol[2] + ks * 32 + lh * 8);
                    load_bfrag(b3n, Wp[half], wcol[3] + ks * 32 + lh * 8);
                }
                acc[0] = __builtin_amdgcn_wmma_f32_16x16x32_bf16(
                    false, a.v, false, b0.v, (short)0, acc[0], false, false);
                acc[1] = __builtin_amdgcn_wmma_f32_16x16x32_bf16(
                    false, a.v, false, b1.v, (short)0, acc[1], false, false);
                acc[2] = __builtin_amdgcn_wmma_f32_16x16x32_bf16(
                    false, a.v, false, b2.v, (short)0, acc[2], false, false);
                acc[3] = __builtin_amdgcn_wmma_f32_16x16x32_bf16(
                    false, a.v, false, b3.v, (short)0, acc[3], false, false);
                if (s < 15) { a = an; b0 = b0n; b1 = b1n; b2 = b2n; b3 = b3n; }
            }

            // epilogue: C layout VGPR r -> M = r + 8*lh, N = lm
            const int row0 = blk << 4;
#pragma unroll
            for (int t = 0; t < 4; ++t) {
                const int c = (cwv * 4 + t) * 16 + lm;
#pragma unroll
                for (int r = 0; r < 8; ++r) {
                    float v = acc[t][r];
                    if (do_relu) v = fmaxf(v, 0.f);
                    const size_t orow = (size_t)(row0 + r + lh * 8);
                    store1(out + orow * DHID + c, v);
                }
            }
        }
        __syncthreads();   // buffer swap: staging of par^1 done, reads of par done
        par ^= 1;
    }
}

static inline int grid_for(int nblk) { return nblk < 512 ? nblk : 512; }

extern "C" void kernel_launch(void* const* d_in, const int* in_sizes, int n_in,
                              void* d_out, int out_size, void* d_ws, size_t ws_size,
                              hipStream_t stream) {
    (void)n_in; (void)out_size; (void)ws_size;
    const float* h0  = (const float*)d_in[0];   // [N, 256]
    const float* h1  = (const float*)d_in[1];   // [N*10, 256]
    const float* h2  = (const float*)d_in[2];   // [N*100, 256]
    const float* w1s = (const float*)d_in[3];
    const float* w1n = (const float*)d_in[4];
    const float* w2s = (const float*)d_in[5];
    const float* w2n = (const float*)d_in[6];

    const int N0 = in_sizes[0] / DHID;   // 10000
    const int N1 = in_sizes[1] / DHID;   // 100000

    // bf16 intermediates in workspace: nh1 [N1,256] then nh0 [N0,256] (~56 MB)
    unsigned short* nh1 = (unsigned short*)d_ws;
    unsigned short* nh0 = nh1 + (size_t)N1 * DHID;
    float* out = (float*)d_out;

    // Layer 1, hop 1 (dominant: streams the 1 GB h2 once, fused mean)
    sage_wmma_kernel<float, unsigned short>
        <<<grid_for(N1 / 16), NTHREADS, 0, stream>>>(h1, h2, w1s, w1n, nh1, N1, 1);
    // Layer 1, hop 0
    sage_wmma_kernel<float, unsigned short>
        <<<grid_for(N0 / 16), NTHREADS, 0, stream>>>(h0, h1, w1s, w1n, nh0, N0, 1);
    // Layer 2 (no activation), reads bf16 intermediates, writes fp32 output
    sage_wmma_kernel<unsigned short, float>
        <<<grid_for(N0 / 16), NTHREADS, 0, stream>>>(nh0, nh1, w2s, w2n, out, N0, 0);
}
